// MoE_30313879175757
// MI455X (gfx1250) — compile-verified, matrix operand order checked by reading, not run on hardware
//
#include <hip/hip_runtime.h>
#include <math.h>

// Problem constants (B=2,S=2048,D=1024,E=8,H=4096,K=2)
#define T_TOK 4096
#define DDIM  1024
#define EEXP  8
#define HDIM  4096

typedef __attribute__((ext_vector_type(16))) __bf16 v16bf;
typedef __attribute__((ext_vector_type(8)))  __bf16 v8bf;
typedef __attribute__((ext_vector_type(8)))  float  v8f;
typedef __attribute__((ext_vector_type(4)))  unsigned int v4u;
typedef __attribute__((ext_vector_type(8)))  int v8i;
typedef __attribute__((ext_vector_type(4)))  int v4i;
typedef __attribute__((ext_vector_type(8)))  unsigned short u16x8;

__device__ __forceinline__ unsigned short f32_to_bf16(float f) {
  unsigned u = __float_as_uint(f);
  unsigned r = u + 0x7FFFu + ((u >> 16) & 1u);   // round-to-nearest-even
  return (unsigned short)(r >> 16);
}

__device__ __forceinline__ v8bf ldg8(const unsigned short* p) {
  return __builtin_bit_cast(v8bf, *reinterpret_cast<const u16x8*>(p));
}
__device__ __forceinline__ v16bf cat16(v8bf lo, v8bf hi) {
  return __builtin_shufflevector(lo, hi, 0,1,2,3,4,5,6,7,8,9,10,11,12,13,14,15);
}

// ---------------- utility kernels ----------------
__global__ void zero_u32(unsigned int* p, long long n) {
  long long i = (long long)blockIdx.x * blockDim.x + threadIdx.x;
  long long s = (long long)gridDim.x * blockDim.x;
  for (; i < n; i += s) p[i] = 0u;
}

__global__ void cvt_bf16(const float* __restrict__ in, unsigned short* __restrict__ out, long long n) {
  long long i = (long long)blockIdx.x * blockDim.x + threadIdx.x;
  long long s = (long long)gridDim.x * blockDim.x;
  for (; i < n; i += s) out[i] = f32_to_bf16(in[i]);
}

// in: [E][R][C] f32  ->  out: [E][C][R] bf16   (LDS-tiled 32x32 transpose)
__global__ void transpose_to_bf16(const float* __restrict__ in, unsigned short* __restrict__ out,
                                  int R, int C) {
  __shared__ float tile[32][33];
  const int e = blockIdx.z;
  const float* inp = in + (size_t)e * R * C;
  unsigned short* outp = out + (size_t)e * R * C;
  const int c0 = blockIdx.x * 32, r0 = blockIdx.y * 32;
  for (int i = threadIdx.y; i < 32; i += 8)
    tile[i][threadIdx.x] = inp[(size_t)(r0 + i) * C + c0 + threadIdx.x];
  __syncthreads();
  for (int i = threadIdx.y; i < 32; i += 8)
    outp[(size_t)(c0 + i) * R + r0 + threadIdx.x] = f32_to_bf16(tile[threadIdx.x][i]);
}

// ---------------- router: logits, softmax, top-2, scatter ----------------
__global__ void router_topk(const float* __restrict__ x, const float* __restrict__ gw,
                            float* __restrict__ logits_out, unsigned int* __restrict__ counts,
                            unsigned short* __restrict__ rows16, float* __restrict__ rowW) {
  const int t = blockIdx.x * 8 + (threadIdx.x >> 5);   // one wave32 per token
  const int lane = threadIdx.x & 31;
  float acc[EEXP];
#pragma unroll
  for (int e = 0; e < EEXP; ++e) acc[e] = 0.f;
  const float* xr = x + (size_t)t * DDIM;
  for (int d = lane; d < DDIM; d += 32) {
    float xv = xr[d];
#pragma unroll
    for (int e = 0; e < EEXP; ++e) acc[e] += xv * gw[e * DDIM + d];
  }
#pragma unroll
  for (int e = 0; e < EEXP; ++e) {
#pragma unroll
    for (int m = 16; m >= 1; m >>= 1) acc[e] += __shfl_xor(acc[e], m, 32);
  }
  if (lane == 0) {
    float mx = acc[0];
#pragma unroll
    for (int e = 1; e < EEXP; ++e) mx = fmaxf(mx, acc[e]);
    float p[EEXP];
#pragma unroll
    for (int e = 0; e < EEXP; ++e) p[e] = expf(acc[e] - mx);
    int i0 = 0;
#pragma unroll
    for (int e = 1; e < EEXP; ++e) if (p[e] > p[i0]) i0 = e;
    int i1 = (i0 == 0) ? 1 : 0;
#pragma unroll
    for (int e = 0; e < EEXP; ++e) if (e != i0 && p[e] > p[i1]) i1 = e;
    float w0 = p[i0], w1 = p[i1], inv = 1.f / (w0 + w1);
    w0 *= inv; w1 *= inv;
#pragma unroll
    for (int e = 0; e < EEXP; ++e) logits_out[t * EEXP + e] = acc[e];
    unsigned s0 = atomicAdd(&counts[i0], 1u);
    rows16[i0 * T_TOK + s0] = (unsigned short)t; rowW[i0 * T_TOK + s0] = w0;
    unsigned s1 = atomicAdd(&counts[i1], 1u);
    rows16[i1 * T_TOK + s1] = (unsigned short)t; rowW[i1 * T_TOK + s1] = w1;
  }
}

// ---------------- grouped-GEMM MoE FFN: gelu(X@W1)@W2, WMMA bf16 ----------------
__global__ __launch_bounds__(256)
void moe_ffn_wmma(const unsigned short* __restrict__ xbf,   // [T][D] bf16
                  const unsigned short* __restrict__ w1t,   // [E][H][D] bf16 (pre-transposed)
                  const unsigned short* __restrict__ w2t,   // [E][D][H] bf16 (pre-transposed)
                  const unsigned int* __restrict__ counts,
                  const unsigned short* __restrict__ rows16,
                  const float* __restrict__ rowW,
                  float* __restrict__ out) {
  __shared__ unsigned short xtile[16 * DDIM];   // 32 KB gathered activations (TDM target)
  __shared__ unsigned short htile[16 * 136];    // padded stride breaks LDS bank conflicts
  __shared__ float s_w[16];
  __shared__ int   s_tok[16];

  const int e  = blockIdx.y;
  const int rt = blockIdx.x;
  const unsigned cnt = counts[e];
  if ((unsigned)(rt * 16) >= cnt) return;       // uniform early-exit of padded tiles

  const int tid  = threadIdx.x;
  const int wv   = tid >> 5;
  const int lane = tid & 31;
  const int lh   = lane >> 4;     // lane half (K sub-block select)
  const int lm   = lane & 15;

  if (tid < 16) {
    const int slot = e * T_TOK + rt * 16 + tid;
    s_tok[tid] = (int)rows16[slot];   // padded slots -> token 0
    s_w[tid]   = rowW[slot];          // padded slots -> weight 0 (harmless)
  }

  // --- TDM gather: 16 scattered token rows (bf16, 1024 elems each) -> xtile ---
  if (wv == 0) {
    const unsigned int* rp =
        (const unsigned int*)(rows16 + (size_t)e * T_TOK + rt * 16);   // 16 u16 idx = 8 dwords
    const unsigned long long ga = (unsigned long long)(size_t)xbf;
    const unsigned lds_base = (unsigned)(size_t)(&xtile[0]);           // LDS offset = addr[31:0]
    v4u g0; v8i g1; v4i g2, g3;
    g0[0] = 0x80000001u;                                   // gather_mode=1, 16b idx, count=1
    g0[1] = (unsigned)__builtin_amdgcn_readfirstlane((int)lds_base);
    g0[2] = (unsigned)(ga & 0xffffffffu);                  // global_addr[31:0]
    g0[3] = (unsigned)((ga >> 32) & 0x1ffffffu) | 0x80000000u;  // addr[56:32] | type=2
    g1[0] = 0x00010000;                                    // data_size = 2 bytes
    g1[1] = (int)(DDIM << 16);                             // tensor_dim0 = 1024 (lo16)
    g1[2] = (int)(T_TOK << 16);                            // tensor_dim1 = 4096 (lo16)
    g1[3] = (int)(DDIM << 16);                             // tile_dim0 = 1024
    g1[4] = 16;                                            // tile_dim1 = #gather indices
    g1[5] = DDIM;                                          // tensor_dim0_stride = 1024
    g1[6] = 0;
    g1[7] = 0;
#pragma unroll
    for (int i = 0; i < 4; ++i) g2[i] = __builtin_amdgcn_readfirstlane((int)rp[i]);
#pragma unroll
    for (int i = 0; i < 4; ++i) g3[i] = __builtin_amdgcn_readfirstlane((int)rp[4 + i]);
    v8i gx = {0, 0, 0, 0, 0, 0, 0, 0};                     // unused trailing D# group (2D gather)
    __builtin_amdgcn_tensor_load_to_lds(g0, g1, g2, g3, gx, 0);
  }
  __builtin_amdgcn_s_wait_tensorcnt(0);
  __syncthreads();

  v8f acc[8];
#pragma unroll
  for (int i = 0; i < 8; ++i) acc[i] = v8f{0.f,0.f,0.f,0.f,0.f,0.f,0.f,0.f};

  const unsigned short* xA = &xtile[lm * DDIM + lh * 8];
  const unsigned short* hA = &htile[lm * 136 + lh * 8];

  for (int c = 0; c < HDIM / 128; ++c) {
    // ---- stage 1: h[:, c*128 + wv*16 .. +16) = X[16x1024] @ W1 slice ----
    v8f h0 = v8f{0.f,0.f,0.f,0.f,0.f,0.f,0.f,0.f};
    v8f h1 = v8f{0.f,0.f,0.f,0.f,0.f,0.f,0.f,0.f};
    const int hcol = c * 128 + wv * 16;
    const unsigned short* bp1 = w1t + ((size_t)e * HDIM + hcol + lm) * DDIM + lh * 8;
#pragma unroll 4
    for (int kk = 0; kk < DDIM; kk += 64) {   // two interleaved accumulator chains
      v16bf a0 = cat16(ldg8(xA + kk), ldg8(xA + kk + 16));
      v16bf b0 = cat16(ldg8(bp1 + kk), ldg8(bp1 + kk + 16));
      h0 = __builtin_amdgcn_wmma_f32_16x16x32_bf16(false, a0, false, b0, (short)0, h0, false, false);
      v16bf a1 = cat16(ldg8(xA + kk + 32), ldg8(xA + kk + 48));
      v16bf b1 = cat16(ldg8(bp1 + kk + 32), ldg8(bp1 + kk + 48));
      h1 = __builtin_amdgcn_wmma_f32_16x16x32_bf16(false, a1, false, b1, (short)0, h1, false, false);
    }
    __syncthreads();                          // previous chunk's htile consumers done
#pragma unroll
    for (int v = 0; v < 8; ++v) {             // exact GELU, C/D layout: row = v + 8*lh
      float hv = h0[v] + h1[v];
      float g = 0.5f * hv * (1.0f + erff(hv * 0.70710678118f));
      htile[(v + 8 * lh) * 136 + wv * 16 + lm] = f32_to_bf16(g);
    }
    __syncthreads();
    // ---- stage 2: acc[16x128] += h_chunk[16x128] @ W2 slice ----
#pragma unroll
    for (int kt = 0; kt < 4; ++kt) {
      v16bf a = cat16(ldg8(hA + kt * 32), ldg8(hA + kt * 32 + 16));
      const unsigned short* bp2 =
          w2t + ((size_t)e * DDIM + wv * 128 + lm) * HDIM + c * 128 + kt * 32 + lh * 8;
#pragma unroll
      for (int nt = 0; nt < 8; ++nt) {        // 8 independent chains -> WMMA ILP
        const unsigned short* bb = bp2 + (size_t)nt * 16 * HDIM;
        v16bf b = cat16(ldg8(bb), ldg8(bb + 16));
        acc[nt] = __builtin_amdgcn_wmma_f32_16x16x32_bf16(false, a, false, b, (short)0, acc[nt], false, false);
      }
    }
  }

  // ---- epilogue: weighted scatter into output rows ----
#pragma unroll
  for (int nt = 0; nt < 8; ++nt) {
    const int col = wv * 128 + nt * 16 + lm;
#pragma unroll
    for (int v = 0; v < 8; ++v) {
      const int m = v + 8 * lh;
      atomicAdd(&out[(size_t)s_tok[m] * DDIM + col], s_w[m] * acc[nt][v]);
    }
  }
}

// ---------------- host-side launch sequence ----------------
extern "C" void kernel_launch(void* const* d_in, const int* in_sizes, int n_in,
                              void* d_out, int out_size, void* d_ws, size_t ws_size,
                              hipStream_t stream) {
  (void)in_sizes; (void)n_in; (void)out_size; (void)ws_size;
  const float* x  = (const float*)d_in[0];
  const float* gw = (const float*)d_in[1];
  const float* w1 = (const float*)d_in[2];
  const float* w2 = (const float*)d_in[3];
  float* out    = (float*)d_out;
  float* logits = out + (size_t)T_TOK * DDIM;

  char* ws = (char*)d_ws;
  unsigned int*   counts = (unsigned int*)ws;                         // 256 B (8 used)
  unsigned short* rows16 = (unsigned short*)(ws + 256);               // 64 KB
  float*          rowW   = (float*)(ws + 256 + 65536);                // 128 KB
  unsigned short* xbf    = (unsigned short*)(ws + 256 + 65536 + 131072);
  unsigned short* w1t    = xbf + (size_t)T_TOK * DDIM;                // [E][H][D] bf16, 64 MB
  unsigned short* w2t    = w1t + (size_t)EEXP * DDIM * HDIM;          // [E][D][H] bf16, 64 MB

  // zero routing metadata (incl. padded-slot weights) + output accumulator
  zero_u32<<<512, 256, 0, stream>>>(counts, (long long)((256 + 65536 + 131072) / 4));
  zero_u32<<<4096, 256, 0, stream>>>((unsigned int*)out, (long long)T_TOK * DDIM);

  // bf16 conversion + weight pre-transpose (weights then live in L2: 128 MB < 192 MB)
  cvt_bf16<<<4096, 256, 0, stream>>>(x, xbf, (long long)T_TOK * DDIM);
  transpose_to_bf16<<<dim3(HDIM / 32, DDIM / 32, EEXP), dim3(32, 8), 0, stream>>>(w1, w1t, DDIM, HDIM);
  transpose_to_bf16<<<dim3(DDIM / 32, HDIM / 32, EEXP), dim3(32, 8), 0, stream>>>(w2, w2t, HDIM, DDIM);

  // routing + grouped GEMM
  router_topk<<<T_TOK / 8, 256, 0, stream>>>(x, gw, logits, counts, rows16, rowW);
  moe_ffn_wmma<<<dim3(T_TOK / 16, EEXP), 256, 0, stream>>>(xbf, w1t, w2t, counts, rows16, rowW, out);
}
